// CasualSelfAttention_16904991277729
// MI455X (gfx1250) — matrix operand order repaired and checked by reading, not compiled, not run on hardware
//
#include <hip/hip_runtime.h>
#include <stdint.h>

// ---------------------------------------------------------------------------
// CDNA5 (gfx1250) causal self-attention:
//   1) gemm_wmma<f32,f32,bf16>   : qkv = x @ w_qkv^T          (8192x3072x1024)
//   2) flash_attn                : att = softmax(QK^T/8)V      (flash, bf16)
//   3) gemm_wmma<bf16,f32,f32>   : out = att @ w_out^T         (8192x1024x1024)
// All matmuls via v_wmma_f32_16x16x32_bf16 (f32 accumulate).
// GEMMs: double-buffered LDS + register prefetch (1 barrier / k-step).
// Attention: K tiles streamed with GLOBAL_LOAD_ASYNC_TO_LDS_B128 (ASYNCcnt)
// when the toolchain exposes the builtin, overlapped with the V transpose.
// ---------------------------------------------------------------------------

typedef __attribute__((ext_vector_type(16))) __bf16 v16bf;
typedef __attribute__((ext_vector_type(8)))  float  v8f;

union FragAB { v16bf v; uint32_t u[8]; };
union FragC  { v8f   v; float    f[8]; };

#if defined(__gfx1250__) &&                                                    \
    __has_builtin(__builtin_amdgcn_global_load_async_to_lds_b128) &&           \
    __has_builtin(__builtin_amdgcn_s_wait_asynccnt)
#define USE_ASYNC_COPY 1
#else
#define USE_ASYNC_COPY 0
#endif

#if USE_ASYNC_COPY
typedef int vi4 __attribute__((vector_size(16)));
typedef __attribute__((address_space(1))) vi4* gvec4p;   // global int4*
typedef __attribute__((address_space(3))) vi4* lvec4p;   // LDS int4*
#endif

static __device__ inline uint16_t f32_to_bf16_bits(float f) {
    uint32_t u = __float_as_uint(f);
    u += 0x7FFFu + ((u >> 16) & 1u);          // round-to-nearest-even
    return (uint16_t)(u >> 16);
}

static __device__ inline v8f wmma_bf16(const FragAB& a, const FragAB& b, v8f c) {
    // (neg_a, A, neg_b, B, c_mod, C, reuse_a, reuse_b)
    return __builtin_amdgcn_wmma_f32_16x16x32_bf16(false, a.v, false, b.v,
                                                   (short)0, c, false, false);
}

// A-matrix 16x32 bf16 fragment (this code's contiguous k order, as load_b_frag):
//   lane 0-15: M=lane,   K-base 0 ; lane 16-31: M=lane-16, K-base 16
//   vgpr i: K = Kbase + 2*i  (pair K,K+1)
static __device__ inline void load_a_frag(FragAB& fr, const uint16_t* tile,
                                          int ldt, int m0, int k0, int lane) {
    const int m  = m0 + (lane & 15);
    const int kb = (lane >> 4) * 16;
    const uint16_t* row = tile + m * ldt + k0;
#pragma unroll
    for (int i = 0; i < 8; ++i) {
        int k = kb + (i << 1);
        fr.u[i] = *(const uint32_t*)(row + k);
    }
}

// B-matrix 32x16 bf16 fragment: lane 0-15 hold K=0..15, lane 16-31 K=16..31,
// vgpr i holds pair (K=Kbase+2i, K+1) for column n = lane&15.
// 'tile' is laid out [n][k] with row stride ldt (so pairs are contiguous u32).
static __device__ inline void load_b_frag(FragAB& fr, const uint16_t* tile,
                                          int ldt, int n0, int k0, int lane) {
    const int n  = n0 + (lane & 15);
    const uint16_t* row = tile + n * ldt + k0 + ((lane >> 4) << 4);
#pragma unroll
    for (int i = 0; i < 8; ++i) fr.u[i] = *(const uint32_t*)(row + 2 * i);
}

// ---- staged (register double-buffer) global->LDS path for the GEMMs -------
template <bool SRC_BF16>
struct Stage { uint4 r[SRC_BF16 ? 2 : 4]; };

template <bool SRC_BF16>
static __device__ inline void fetch16(Stage<SRC_BF16>& s, const void* base,
                                      size_t elemoff) {
    if constexpr (SRC_BF16) {
        const uint4* p = (const uint4*)((const uint16_t*)base + elemoff);
        s.r[0] = p[0]; s.r[1] = p[1];
    } else {
        const uint4* p = (const uint4*)((const float*)base + elemoff);
        s.r[0] = p[0]; s.r[1] = p[1]; s.r[2] = p[2]; s.r[3] = p[3];
    }
}

static __device__ inline uint32_t pack_bf16(uint32_t lo, uint32_t hi) {
    return (uint32_t)f32_to_bf16_bits(__uint_as_float(lo)) |
           ((uint32_t)f32_to_bf16_bits(__uint_as_float(hi)) << 16);
}

template <bool SRC_BF16>
static __device__ inline void stage_store16(uint32_t* dst,
                                            const Stage<SRC_BF16>& s) {
    if constexpr (SRC_BF16) {
        dst[0] = s.r[0].x; dst[1] = s.r[0].y; dst[2] = s.r[0].z; dst[3] = s.r[0].w;
        dst[4] = s.r[1].x; dst[5] = s.r[1].y; dst[6] = s.r[1].z; dst[7] = s.r[1].w;
    } else {
#pragma unroll
        for (int q = 0; q < 4; ++q) {
            dst[2 * q]     = pack_bf16(s.r[q].x, s.r[q].y);
            dst[2 * q + 1] = pack_bf16(s.r[q].z, s.r[q].w);
        }
    }
}

// ---- async (or manual) 64-byte row copy for bf16 tiles in attention -------
static __device__ inline void copy_row64(uint16_t* ldst, const uint16_t* gsrc) {
#if USE_ASYNC_COPY
    gvec4p g = (gvec4p)gsrc;
    lvec4p l = (lvec4p)ldst;
    __builtin_amdgcn_global_load_async_to_lds_b128(g, l, 0, 0);
    __builtin_amdgcn_global_load_async_to_lds_b128(g, l, 16, 0);
    __builtin_amdgcn_global_load_async_to_lds_b128(g, l, 32, 0);
    __builtin_amdgcn_global_load_async_to_lds_b128(g, l, 48, 0);
#else
    const uint4* p = (const uint4*)gsrc;
    uint32_t* d = (uint32_t*)ldst;
    uint4 a = p[0], b = p[1], c = p[2], e = p[3];
    d[0]=a.x; d[1]=a.y; d[2]=a.z; d[3]=a.w;
    d[4]=b.x; d[5]=b.y; d[6]=b.z; d[7]=b.w;
    d[8]=c.x; d[9]=c.y; d[10]=c.z; d[11]=c.w;
    d[12]=e.x; d[13]=e.y; d[14]=e.z; d[15]=e.w;
#endif
}

static __device__ inline void async_drain() {
#if USE_ASYNC_COPY
    __builtin_amdgcn_s_wait_asynccnt(0);
#endif
}

// ---------------------------------------------------------------------------
// GEMM: Out[M,N] = A[M,K] @ B[N,K]^T  (weights stored (out,in) as in nn.Linear)
// 256 threads = 8 waves; block tile 128x128, k-step 32; wave tile 32x64.
// Double-buffered LDS; next tile prefetched into VGPRs during the WMMAs.
// ---------------------------------------------------------------------------
template <bool A_BF16, bool B_BF16, bool OUT_BF16>
__global__ __launch_bounds__(256)
void gemm_wmma(const void* __restrict__ Ap, const void* __restrict__ Bp,
               void* __restrict__ Out, int M, int N, int K) {
    constexpr int LDT = 36;                       // 32 + pad (keeps u32 align)
    __shared__ uint16_t As[2][128 * LDT];
    __shared__ uint16_t Bs[2][128 * LDT];

    const int lane = threadIdx.x & 31;
    const int wave = threadIdx.x >> 5;
    const int wm   = (wave >> 1) * 32;            // 0,32,64,96
    const int wn   = (wave & 1) * 64;             // 0,64
    const int m0   = blockIdx.y * 128;
    const int n0   = blockIdx.x * 128;

    const int r  = threadIdx.x >> 1;              // 0..127
    const int cs = (threadIdx.x & 1) * 16;        // 0 or 16
    const size_t aoff = (size_t)(m0 + r) * K + cs;
    const size_t boff = (size_t)(n0 + r) * K + cs;

    FragC acc[2][4];
#pragma unroll
    for (int mi = 0; mi < 2; ++mi)
#pragma unroll
        for (int j = 0; j < 4; ++j)
#pragma unroll
            for (int i = 0; i < 8; ++i) acc[mi][j].f[i] = 0.0f;

    Stage<A_BF16> sa; Stage<B_BF16> sb;
    fetch16<A_BF16>(sa, Ap, aoff);
    fetch16<B_BF16>(sb, Bp, boff);
    stage_store16<A_BF16>((uint32_t*)&As[0][r * LDT + cs], sa);
    stage_store16<B_BF16>((uint32_t*)&Bs[0][r * LDT + cs], sb);
    __syncthreads();

    const int nk = K >> 5;
    for (int kt = 0; kt < nk; ++kt) {
        const int cur = kt & 1;
        if (kt + 1 < nk) {                        // prefetch next tile -> regs
            fetch16<A_BF16>(sa, Ap, aoff + ((kt + 1) << 5));
            fetch16<B_BF16>(sb, Bp, boff + ((kt + 1) << 5));
        }

        FragAB a[2], b[4];
#pragma unroll
        for (int mi = 0; mi < 2; ++mi)
            load_a_frag(a[mi], &As[cur][0], LDT, wm + 16 * mi, 0, lane);
#pragma unroll
        for (int j = 0; j < 4; ++j)
            load_b_frag(b[j], &Bs[cur][0], LDT, wn + 16 * j, 0, lane);
#pragma unroll
        for (int mi = 0; mi < 2; ++mi)
#pragma unroll
            for (int j = 0; j < 4; ++j)
                acc[mi][j].v = wmma_bf16(a[mi], b[j], acc[mi][j].v);

        if (kt + 1 < nk) {                        // regs -> other LDS buffer
            stage_store16<A_BF16>((uint32_t*)&As[cur ^ 1][r * LDT + cs], sa);
            stage_store16<B_BF16>((uint32_t*)&Bs[cur ^ 1][r * LDT + cs], sb);
        }
        __syncthreads();
    }

    // C/D layout: col n = lane&15, row m = i + 8*(lane>>4)
    const int cn = lane & 15;
    const int rm = (lane >> 4) * 8;
#pragma unroll
    for (int mi = 0; mi < 2; ++mi)
#pragma unroll
        for (int j = 0; j < 4; ++j)
#pragma unroll
            for (int i = 0; i < 8; ++i) {
                size_t row = (size_t)(m0 + wm + 16 * mi + rm + i);
                size_t col = (size_t)(n0 + wn + 16 * j + cn);
                if constexpr (OUT_BF16)
                    ((uint16_t*)Out)[row * N + col] = f32_to_bf16_bits(acc[mi][j].f[i]);
                else
                    ((float*)Out)[row * N + col] = acc[mi][j].f[i];
            }
}

// ---------------------------------------------------------------------------
// Flash attention: one block per (b, h, 64-query tile); 128 threads = 4 waves,
// each wave owns 16 query rows. Online softmax, causal mask, kv tiles of 64.
// qkv layout: (B, T, 3C) bf16 ; att layout: (B, T, C) bf16.
// K tile streamed via async LDS copy, overlapped with the V transpose.
// ---------------------------------------------------------------------------
__global__ __launch_bounds__(128)
void flash_attn(const uint16_t* __restrict__ qkv, uint16_t* __restrict__ att) {
    constexpr int T = 2048, C3 = 3072, C = 1024;
    constexpr int LDT = 72;                       // 64 + pad
    __shared__ uint16_t Qs[64 * LDT];
    __shared__ uint16_t Ks[64 * LDT];
    __shared__ uint16_t Vt[64 * LDT];             // transposed: [hd][kv]
    __shared__ uint16_t Ps[64 * LDT];
    __shared__ float    Ss[64 * 68];
    __shared__ float    rowm[64], rowl[64], rowscale[64];

    const int qt = blockIdx.x & 31;               // T/64 = 32 query tiles
    const int h  = (blockIdx.x >> 5) & 15;
    const int b  = blockIdx.x >> 9;
    const int q0 = qt * 64;

    const int tid = threadIdx.x, lane = tid & 31, wave = tid >> 5;
    const int r = tid & 63, half = tid >> 6;      // tile-load coords

    const size_t base = (size_t)b * T * C3 + (size_t)h * 64;

    // Q tile -> LDS (async if available)
    copy_row64(&Qs[r * LDT + half * 32],
               qkv + base + (size_t)(q0 + r) * C3 + half * 32);
    if (tid < 64) { rowm[tid] = -3.0e30f; rowl[tid] = 0.0f; }
    async_drain();
    __syncthreads();

    FragAB aQ[2];
    load_a_frag(aQ[0], Qs, LDT, wave * 16, 0, lane);
    load_a_frag(aQ[1], Qs, LDT, wave * 16, 32, lane);

    FragC accO[4];
#pragma unroll
    for (int j = 0; j < 4; ++j)
#pragma unroll
        for (int i = 0; i < 8; ++i) accO[j].f[i] = 0.0f;

    const int ntiles = qt + 1;                    // causal: kv <= q
    for (int t = 0; t < ntiles; ++t) {
        const int kv0 = t * 64;
        __syncthreads();
        // K tile: async copy runs while we do the V transpose below
        copy_row64(&Ks[r * LDT + half * 32],
                   qkv + base + (size_t)C + (size_t)(kv0 + r) * C3 + half * 32);
        { // V tile, transposed into Vt[hd][kv]
            const uint16_t* vs = qkv + base + (size_t)(2 * C) +
                                 (size_t)(kv0 + r) * C3 + half * 32;
#pragma unroll
            for (int c = 0; c < 32; ++c) Vt[(half * 32 + c) * LDT + r] = vs[c];
        }
        async_drain();
        __syncthreads();

        // S = Q K^T : B-frag element (k=hd, n=kv) = Ks[kv][hd] -> [n][k] layout
        FragC s[4];
#pragma unroll
        for (int j = 0; j < 4; ++j) {
#pragma unroll
            for (int i = 0; i < 8; ++i) s[j].f[i] = 0.0f;
            FragAB bk;
            load_b_frag(bk, Ks, LDT, 16 * j, 0, lane);
            s[j].v = wmma_bf16(aQ[0], bk, s[j].v);
            load_b_frag(bk, Ks, LDT, 16 * j, 32, lane);
            s[j].v = wmma_bf16(aQ[1], bk, s[j].v);
        }

        // scale (1/sqrt(64)=0.125), causal mask, spill to LDS for softmax
        const int cn = lane & 15, rm = (lane >> 4) * 8;
#pragma unroll
        for (int j = 0; j < 4; ++j)
#pragma unroll
            for (int i = 0; i < 8; ++i) {
                int qg = q0 + wave * 16 + rm + i;
                int kg = kv0 + 16 * j + cn;
                float v = s[j].f[i] * 0.125f;
                if (kg > qg) v = -3.0e30f;
                Ss[(wave * 16 + rm + i) * 68 + 16 * j + cn] = v;
            }

        // online softmax: 2 lanes per row, 32 cols each (same-wave LDS in order)
        const int srow = wave * 16 + (lane >> 1);
        const int c0   = (lane & 1) * 32;
        const float* sp = &Ss[srow * 68 + c0];
        float pmax = -3.0e30f;
#pragma unroll
        for (int c = 0; c < 32; ++c) pmax = fmaxf(pmax, sp[c]);
        float omax = fmaxf(pmax, __shfl_xor(pmax, 1, 32));
        float mold = rowm[srow];
        float mnew = fmaxf(mold, omax);
        float alpha = __expf(mold - mnew);
        float psum = 0.0f;
        uint16_t* pp = &Ps[srow * LDT + c0];
#pragma unroll
        for (int c = 0; c < 32; ++c) {
            float e = __expf(sp[c] - mnew);
            pp[c] = f32_to_bf16_bits(e);
            psum += e;
        }
        float lsum = psum + __shfl_xor(psum, 1, 32);
        rowl[srow] = alpha * rowl[srow] + lsum;   // both lanes write same value
        rowm[srow] = mnew;
        rowscale[srow] = alpha;

        // rescale running accumulator by per-row alpha
        float al[8];
#pragma unroll
        for (int i = 0; i < 8; ++i) al[i] = rowscale[wave * 16 + rm + i];
#pragma unroll
        for (int j = 0; j < 4; ++j)
#pragma unroll
            for (int i = 0; i < 8; ++i) accO[j].f[i] *= al[i];

        // accO += P V : B-frag element (k=kv, n=hd) = Vt[hd][kv] -> [n][k]
        FragAB aP[2];
        load_a_frag(aP[0], Ps, LDT, wave * 16, 0, lane);
        load_a_frag(aP[1], Ps, LDT, wave * 16, 32, lane);
#pragma unroll
        for (int j = 0; j < 4; ++j) {
            FragAB bv;
            load_b_frag(bv, Vt, LDT, 16 * j, 0, lane);
            accO[j].v = wmma_bf16(aP[0], bv, accO[j].v);
            load_b_frag(bv, Vt, LDT, 16 * j, 32, lane);
            accO[j].v = wmma_bf16(aP[1], bv, accO[j].v);
        }
    }

    // normalize by l and store att (bf16, (B,T,C), heads re-interleaved)
    const int cn = lane & 15, rm = (lane >> 4) * 8;
    float rl[8];
#pragma unroll
    for (int i = 0; i < 8; ++i) rl[i] = 1.0f / rowl[wave * 16 + rm + i];
#pragma unroll
    for (int j = 0; j < 4; ++j)
#pragma unroll
        for (int i = 0; i < 8; ++i) {
            size_t row = (size_t)(b * T + q0 + wave * 16 + rm + i);
            size_t col = (size_t)(h * 64 + 16 * j + cn);
            att[row * C + col] = f32_to_bf16_bits(accO[j].f[i] * rl[i]);
        }
}

// ---------------------------------------------------------------------------
extern "C" void kernel_launch(void* const* d_in, const int* in_sizes, int n_in,
                              void* d_out, int out_size, void* d_ws, size_t ws_size,
                              hipStream_t stream) {
    (void)in_sizes; (void)n_in; (void)out_size; (void)ws_size;
    const float* x     = (const float*)d_in[0];   // (4,2048,1024)
    const float* w_qkv = (const float*)d_in[1];   // (3072,1024)
    const float* w_out = (const float*)d_in[2];   // (1024,1024)
    float* out = (float*)d_out;                   // (4,2048,1024) f32

    const int M = 4 * 2048;                       // 8192
    uint16_t* qkvb = (uint16_t*)d_ws;             // (8192,3072) bf16 : 48 MB
    uint16_t* attb = qkvb + (size_t)M * 3072;     // (8192,1024) bf16 : 16 MB

    // 1) qkv = x @ w_qkv^T  (f32 in, bf16 out)
    gemm_wmma<false, false, true><<<dim3(3072 / 128, M / 128), 256, 0, stream>>>(
        x, w_qkv, qkvb, M, 3072, 1024);

    // 2) flash attention over (b=4, h=16, 32 query tiles of 64)
    flash_attn<<<dim3(4 * 16 * 32), 128, 0, stream>>>(qkvb, attb);

    // 3) out = att @ w_out^T  (bf16 A, f32 B, f32 out)
    gemm_wmma<true, false, false><<<dim3(1024 / 128, M / 128), 256, 0, stream>>>(
        attb, w_out, out, M, 1024, 1024);
}
